// SparseMoE_61813169324591
// MI455X (gfx1250) — matrix operand order repaired and checked by reading, not compiled, not run on hardware
//
#include <hip/hip_runtime.h>

// ---------------- types for WMMA ----------------
typedef __attribute__((ext_vector_type(16))) __bf16 v16bf;
typedef __attribute__((ext_vector_type(8)))  float  v8f;

union ABU {
  v16bf v;
  uint4 q[2];
};

// Sizes (fixed by the reference)
#define NB 4            // batch
#define SS 1024         // seq
#define DD 1024         // d_model
#define FF 4096         // d_ff
#define EE 8            // experts
#define NTOK (NB*SS)    // 4096 tokens

#define MT 32           // token tile (two 16-row WMMA blocks)
#define FC 128          // F-chunk width held in LDS
#define XS_STRIDE 1032  // padded row stride (shorts): 2064B = 129*16, +4 banks/row
#define HC_STRIDE 136   // padded row stride (shorts):  272B =  17*16, +4 banks/row

// float -> bf16 round-to-nearest-even
__device__ __forceinline__ unsigned short f2bf(float f) {
  unsigned int u = __float_as_uint(f);
  unsigned int r = u + 0x7FFFu + ((u >> 16) & 1u);
  return (unsigned short)(r >> 16);
}

// ---------------------------------------------------------------------------
// Kernel 1: convert+swizzle a weight matrix [nExp][Kdim][Ndim] (fp32, row major)
// into the CDNA5 WMMA bf16 B-operand layout, blocked as
//   blk = (e*(Ndim/16)+nb)*(Kdim/32)+kb,  within blk: [lane(32)][vgpr(8) pairs]
// B layout (32x16 bf16): lane L holds col N=L%16; VGPR v holds K=2v (+8 if L>=16, +8 if v>=4).
// ---------------------------------------------------------------------------
__global__ __launch_bounds__(256) void swizzle_convert_kernel(
    const float* __restrict__ W, unsigned int* __restrict__ dst,
    int Kdim, int Ndim, long total_pairs) {
  long idx = (long)blockIdx.x * blockDim.x + threadIdx.x;
  if (idx >= total_pairs) return;
  int pairInBlk = (int)(idx & 255);
  int v = pairInBlk & 7;
  int lane = pairInBlk >> 3;
  long blk = idx >> 8;
  int kblocks = Kdim >> 5;
  int kb = (int)(blk % kblocks);
  long t = blk / kblocks;
  int nb = (int)(t % (Ndim >> 4));
  int e  = (int)(t / (Ndim >> 4));
  int k = kb * 32 + 2 * v + ((lane & 16) ? 8 : 0) + ((v & 4) ? 8 : 0);
  int n = nb * 16 + (lane & 15);
  const float* s = W + ((long)e * Kdim + k) * (long)Ndim + n;
  unsigned short lo = f2bf(s[0]);
  unsigned short hi = f2bf(s[Ndim]);     // K=k+1, same column
  dst[idx] = ((unsigned int)hi << 16) | (unsigned int)lo;
}

// ---------------------------------------------------------------------------
// Kernel 2: router. One wave (32 lanes) per token, 8 waves / block.
// ---------------------------------------------------------------------------
__global__ __launch_bounds__(256) void router_kernel(
    const float* __restrict__ x, const float* __restrict__ gate_w,
    int* __restrict__ list, float* __restrict__ wslot,
    int* __restrict__ counts, float* __restrict__ partials) {
  int wv = threadIdx.x >> 5;
  int lane = threadIdx.x & 31;
  int token = blockIdx.x * 8 + wv;

  float acc[EE];
#pragma unroll
  for (int e = 0; e < EE; ++e) acc[e] = 0.0f;

  const float* xr = x + (long)token * DD;
  for (int d = lane; d < DD; d += 32) {
    float xv = xr[d];
    const float* g = gate_w + d * EE;
#pragma unroll
    for (int e = 0; e < EE; ++e) acc[e] += xv * g[e];
  }
  // wave32 butterfly reduce (all lanes end with full sums, bitwise identical)
#pragma unroll
  for (int m = 16; m >= 1; m >>= 1) {
#pragma unroll
    for (int e = 0; e < EE; ++e) acc[e] += __shfl_xor(acc[e], m, 32);
  }

  __shared__ float sp[8][EE];
  if (lane == 0) {
    float mx = acc[0];
#pragma unroll
    for (int e = 1; e < EE; ++e) mx = fmaxf(mx, acc[e]);
    float pr[EE]; float sum = 0.0f;
#pragma unroll
    for (int e = 0; e < EE; ++e) { pr[e] = expf(acc[e] - mx); sum += pr[e]; }
    float inv = 1.0f / sum;
#pragma unroll
    for (int e = 0; e < EE; ++e) { pr[e] *= inv; sp[wv][e] = pr[e]; }
    // top-2 (ties -> lowest index, matching jax top_k)
    int e0 = 0;
#pragma unroll
    for (int e = 1; e < EE; ++e) if (pr[e] > pr[e0]) e0 = e;
    int e1 = (e0 == 0) ? 1 : 0;
#pragma unroll
    for (int e = 0; e < EE; ++e) if (e != e0 && pr[e] > pr[e1]) e1 = e;
    float w0 = pr[e0], w1 = pr[e1];
    float wsum = w0 + w1;
    w0 /= wsum; w1 /= wsum;
    int i0 = atomicAdd(&counts[e0], 1);
    int i1 = atomicAdd(&counts[e1], 1);
    list[e0 * NTOK + i0] = token * 2;
    list[e1 * NTOK + i1] = token * 2 + 1;
    wslot[token * 2]     = w0;
    wslot[token * 2 + 1] = w1;
  }
  __syncthreads();
  if (threadIdx.x < EE) {
    float s = 0.0f;
#pragma unroll
    for (int w = 0; w < 8; ++w) s += sp[w][threadIdx.x];   // fixed order
    partials[blockIdx.x * EE + threadIdx.x] = s;
  }
}

// ---------------------------------------------------------------------------
// Kernel 3: fused expert FFN for a 32-token tile of one expert.
// Loops over F in chunks of 128:
//   stage A: Hch = relu(X @ Wup[:,chunk] + b_up[chunk])  (32 x 128 bf16, LDS)
//   stage B: Yacc += Hch @ Wdown[chunk,:]                (32 x 1024 fp32, VGPRs)
// Each B tile feeds 2 WMMAs (both row blocks); each stage-B A tile feeds 8.
// ---------------------------------------------------------------------------
__global__ __launch_bounds__(256) void ffn_kernel(
    const float* __restrict__ x,
    const unsigned int* __restrict__ wupz,
    const unsigned int* __restrict__ wdnz,
    const float* __restrict__ b_up, const float* __restrict__ b_down,
    const int* __restrict__ list, const int* __restrict__ counts,
    float* __restrict__ Y) {
  __shared__ __align__(16) unsigned short Xs[MT * XS_STRIDE];   // ~64.5 KB
  __shared__ __align__(16) unsigned short Hch[MT * HC_STRIDE];  // ~8.5 KB

  int e = blockIdx.x >> 7;        // 128 tiles reserved per expert
  int t = blockIdx.x & 127;
  int cnt = counts[e];
  if (t * MT >= cnt) return;

  int tid  = threadIdx.x;
  int wv   = tid >> 5;
  int lane = tid & 31;

  // per-lane code for token row `lane` of this tile (same in every wave)
  int i = t * MT + lane;
  int mycode = (i < cnt) ? list[e * NTOK + i] : -1;

  // gather 32 token rows of X into LDS as bf16 (zeros for pad rows)
  for (int j = tid; j < MT * DD; j += 256) {
    int m = j >> 10, d = j & (DD - 1);
    int c = __shfl(mycode, m, 32);    // m is uniform across the wave here
    float v = (c >= 0) ? x[(long)(c >> 1) * DD + d] : 0.0f;
    Xs[m * XS_STRIDE + d] = f2bf(v);
  }

  int mrow = (lane & 16) ? 8 : 0;   // C/D layout: M = r (+8 for hi lanes)
  int col  = lane & 15;             // B/C/D layout: N = lane % 16
  int aoff = (lane & 16) ? 8 : 0;   // A layout: hi lanes start at K+8

  const unsigned short* xrow0 = Xs + (lane & 15) * XS_STRIDE;        // rows 0..15
  const unsigned short* xrow1 = xrow0 + 16 * XS_STRIDE;              // rows 16..31
  const unsigned short* hrow0 = Hch + (lane & 15) * HC_STRIDE;
  const unsigned short* hrow1 = hrow0 + 16 * HC_STRIDE;

  // persistent Y accumulators: [ci]=rows 0..15, [8+ci]=rows 16..31, col tile wv+8*ci
  v8f acc[16];
#pragma unroll
  for (int q = 0; q < 16; ++q) acc[q] = {};

  float bu_base;
  for (int fc = 0; fc < FF / FC; ++fc) {
    __syncthreads();   // Hch free (prev stage B done), Xs ready on first iter

    // ---- stage A: this wave computes col-tile `wv` of the 128-wide chunk ----
    int nbg = fc * (FC / 16) + wv;                    // global F col-tile
    v8f ca0 = {}, ca1 = {};
    long blkbase = ((long)e * (FF / 16) + nbg) * (DD / 32);
#pragma unroll 4
    for (int kb = 0; kb < DD / 32; ++kb) {
      ABU a0, a1, b;
      const uint4* bp = ((const uint4*)wupz) + (blkbase + kb) * 64 + lane * 2;
      b.q[0] = bp[0];
      b.q[1] = bp[1];
      const uint4* ap0 = (const uint4*)(xrow0 + kb * 32 + aoff);
      a0.q[0] = ap0[0]; a0.q[1] = ap0[2];
      const uint4* ap1 = (const uint4*)(xrow1 + kb * 32 + aoff);
      a1.q[0] = ap1[0]; a1.q[1] = ap1[2];
      ca0 = __builtin_amdgcn_wmma_f32_16x16x32_bf16(false, a0.v, false, b.v,
                                                    (short)0, ca0, false, false);
      ca1 = __builtin_amdgcn_wmma_f32_16x16x32_bf16(false, a1.v, false, b.v,
                                                    (short)0, ca1, false, false);
    }
    bu_base = b_up[(long)e * FF + nbg * 16 + col];
#pragma unroll
    for (int r = 0; r < 8; ++r) {
      float h0 = fmaxf(ca0[r] + bu_base, 0.0f);
      float h1 = fmaxf(ca1[r] + bu_base, 0.0f);
      Hch[(r + mrow) * HC_STRIDE + wv * 16 + col]        = f2bf(h0);
      Hch[(16 + r + mrow) * HC_STRIDE + wv * 16 + col]   = f2bf(h1);
    }
    __syncthreads();   // Hch complete

    // ---- stage B: accumulate Y += Hch @ Wdown[chunk,:] ----
#pragma unroll
    for (int kb2 = 0; kb2 < FC / 32; ++kb2) {
      ABU a0, a1;
      const uint4* hp0 = (const uint4*)(hrow0 + kb2 * 32 + aoff);
      a0.q[0] = hp0[0]; a0.q[1] = hp0[2];
      const uint4* hp1 = (const uint4*)(hrow1 + kb2 * 32 + aoff);
      a1.q[0] = hp1[0]; a1.q[1] = hp1[2];
#pragma unroll
      for (int ci = 0; ci < 8; ++ci) {
        int nb2 = wv + ci * 8;                        // output D col-tile
        long blk2 = ((long)e * (DD / 16) + nb2) * (FF / 32) + fc * (FC / 32) + kb2;
        ABU b;
        const uint4* bp = ((const uint4*)wdnz) + blk2 * 64 + lane * 2;
        b.q[0] = bp[0];
        b.q[1] = bp[1];
        acc[ci] = __builtin_amdgcn_wmma_f32_16x16x32_bf16(
            false, a0.v, false, b.v, (short)0, acc[ci], false, false);
        acc[8 + ci] = __builtin_amdgcn_wmma_f32_16x16x32_bf16(
            false, a1.v, false, b.v, (short)0, acc[8 + ci], false, false);
      }
    }
  }

  // ---- epilogue: add b_down, scatter rows to per-(token,slot) buffer ----
#pragma unroll
  for (int ci = 0; ci < 8; ++ci) {
    int nb2 = wv + ci * 8;
    float bd = b_down[(long)e * DD + nb2 * 16 + col];
#pragma unroll
    for (int rb = 0; rb < 2; ++rb) {
#pragma unroll
      for (int r = 0; r < 8; ++r) {
        int row = rb * 16 + r + mrow;
        int cd = __shfl(mycode, row, 32);   // ds_bpermute (non-uniform ok)
        if (cd >= 0)
          Y[(long)cd * DD + nb2 * 16 + col] = acc[rb * 8 + ci][r] + bd;
      }
    }
  }
}

// ---------------------------------------------------------------------------
// Kernel 4: combine slots in fixed order (bitwise deterministic).
// ---------------------------------------------------------------------------
__global__ __launch_bounds__(256) void combine_kernel(
    const float* __restrict__ Y, const float* __restrict__ wslot,
    float* __restrict__ out) {
  long j = (long)blockIdx.x * blockDim.x + threadIdx.x;
  if (j >= (long)NTOK * DD) return;
  long n = j >> 10;
  long d = j & (DD - 1);
  out[j] = wslot[2 * n]     * Y[(2 * n) * (long)DD + d] +
           wslot[2 * n + 1] * Y[(2 * n + 1) * (long)DD + d];
}

// ---------------------------------------------------------------------------
// Kernel 5: aux load-balancing loss, fixed-order reduction.
// ---------------------------------------------------------------------------
__global__ void aux_kernel(const int* __restrict__ counts,
                           const float* __restrict__ partials,
                           float* __restrict__ out) {
  if (threadIdx.x == 0 && blockIdx.x == 0) {
    float s = 0.0f;
    for (int e = 0; e < EE; ++e) {
      float imp = 0.0f;
      for (int b = 0; b < NTOK / 8; ++b) imp += partials[b * EE + e];
      float load = (float)counts[e] / (float)NTOK;
      s += load * (imp / (float)NTOK);
    }
    out[(long)NTOK * DD] = (float)EE * s * 0.001f;   // E * sum * Z_COEF
  }
}

// ---------------------------------------------------------------------------
extern "C" void kernel_launch(void* const* d_in, const int* in_sizes, int n_in,
                              void* d_out, int out_size, void* d_ws, size_t ws_size,
                              hipStream_t stream) {
  const float* x      = (const float*)d_in[0];
  const float* gate_w = (const float*)d_in[1];
  const float* w_up   = (const float*)d_in[2];
  const float* b_up   = (const float*)d_in[3];
  const float* w_down = (const float*)d_in[4];
  const float* b_down = (const float*)d_in[5];
  float* out = (float*)d_out;

  char* ws = (char*)d_ws;
  size_t off = 0;
  auto alloc = [&](size_t bytes) -> char* {
    char* p = ws + off;
    off += (bytes + 255) & ~(size_t)255;
    return p;
  };
  unsigned int* wupz   = (unsigned int*)alloc((size_t)EE * DD * FF * 2);  // 64 MB bf16
  unsigned int* wdnz   = (unsigned int*)alloc((size_t)EE * FF * DD * 2);  // 64 MB bf16
  float*        Y      = (float*)alloc((size_t)NTOK * 2 * DD * 4);        // 32 MB
  int*          list   = (int*)alloc((size_t)EE * NTOK * 4);
  float*        wslot  = (float*)alloc((size_t)NTOK * 2 * 4);
  int*          counts = (int*)alloc(EE * sizeof(int));
  float*        partials = (float*)alloc((size_t)(NTOK / 8) * EE * 4);
  (void)in_sizes; (void)n_in; (void)out_size; (void)ws_size;

  hipMemsetAsync(counts, 0, EE * sizeof(int), stream);

  long up_pairs = (long)EE * DD * FF / 2;   // 16,777,216
  long dn_pairs = (long)EE * FF * DD / 2;
  swizzle_convert_kernel<<<(int)((up_pairs + 255) / 256), 256, 0, stream>>>(
      w_up, wupz, DD, FF, up_pairs);
  swizzle_convert_kernel<<<(int)((dn_pairs + 255) / 256), 256, 0, stream>>>(
      w_down, wdnz, FF, DD, dn_pairs);

  router_kernel<<<NTOK / 8, 256, 0, stream>>>(x, gate_w, list, wslot, counts,
                                              partials);

  ffn_kernel<<<EE * 128, 256, 0, stream>>>(x, wupz, wdnz, b_up, b_down, list,
                                           counts, Y);

  long outelems = (long)NTOK * DD;
  combine_kernel<<<(int)((outelems + 255) / 256), 256, 0, stream>>>(Y, wslot, out);

  aux_kernel<<<1, 32, 0, stream>>>(counts, partials, out);
}
